// GNBlock_44126493999528
// MI455X (gfx1250) — matrix operand order repaired and checked by reading, not compiled, not run on hardware
//
#include <hip/hip_runtime.h>
#include <hip/hip_bf16.h>

#define DD 128
#define NEDGES 320000
#define NNODES 20000

typedef __attribute__((ext_vector_type(16))) __bf16        v16bf;
typedef __attribute__((ext_vector_type(8)))  float         v8f;
typedef __attribute__((ext_vector_type(8)))  unsigned int  v8u;
typedef __attribute__((ext_vector_type(4)))  unsigned int  u32x4;
typedef __attribute__((ext_vector_type(8)))  int           i32x8;
typedef __attribute__((ext_vector_type(4)))  int           i32x4;

#if __has_builtin(__builtin_amdgcn_tensor_load_to_lds) && __has_builtin(__builtin_amdgcn_s_wait_tensorcnt)
#define USE_TDM 1
#else
#define USE_TDM 0
#endif

__device__ __forceinline__ unsigned pack_bf16x2(float a, float b) {
  union { float f; unsigned u; } ua, ub;
  ua.f = a; ub.f = b;
  unsigned x = (ua.u + 0x7FFFu + ((ua.u >> 16) & 1u)) >> 16;
  unsigned y = (ub.u + 0x7FFFu + ((ub.u >> 16) & 1u)) >> 16;
  return (y << 16) | (x & 0xFFFFu);
}

__device__ __forceinline__ v8f wmma_bf16(v16bf a, v16bf b, v8f c) {
  // D = A(16x32 bf16) x B(32x16 bf16) + C(16x16 f32)
  return __builtin_amdgcn_wmma_f32_16x16x32_bf16(
      /*neg_a=*/false, a, /*neg_b=*/false, b,
      /*c_mod=*/(short)0, c, /*reuse_a=*/false, /*reuse_b=*/false);
}

#if USE_TDM
// Stage a contiguous 1-D block (n8 x 8-byte elements) from global into LDS
// via the Tensor Data Mover.  D# per CDNA5 ISA:
//   group0: count=1, lds_addr, global_addr[56:0], type=2
//   group1: data_size=3 (8B), tensor_dim0=n8, tile_dim0=n8, dim0_stride=n8
// Caller: issue from ONE wave, then workgroup barrier.
__device__ __forceinline__ void tdm_stage_1d(const void* gsrc, unsigned lds_off,
                                             unsigned n8) {
  unsigned long long ga = (unsigned long long)(size_t)gsrc;
  u32x4 g0;
  g0[0] = 1u;                                    // count=1 (valid user D#)
  g0[1] = lds_off;                               // lds_addr
  g0[2] = (unsigned)ga;                          // global_addr[31:0]
  g0[3] = (unsigned)((ga >> 32) & 0x01FFFFFFu)   // global_addr[56:32]
        | (2u << 30);                            // type=2 ("image")
  i32x8 g1;
  g1[0] = (int)(3u << 16);                       // data_size=3 (8 bytes)
  g1[1] = (int)((n8 & 0xFFFFu) << 16);           // tensor_dim0[15:0] @ bits 63:48
  g1[2] = (int)(n8 >> 16);                       // tensor_dim0[31:16] @ bits 79:64
  g1[3] = (int)((n8 & 0xFFFFu) << 16);           // tile_dim0 @ bits 127:112
  g1[4] = 0;                                     // tile_dim1/2 unused
  g1[5] = (int)n8;                               // tensor_dim0_stride[31:0]
  g1[6] = 0;
  g1[7] = 0;
  i32x4 z4 = {0, 0, 0, 0};
#if defined(__clang_major__) && (__clang_major__ >= 23)
  i32x8 z8 = {0, 0, 0, 0, 0, 0, 0, 0};
  __builtin_amdgcn_tensor_load_to_lds(g0, g1, z4, z4, z8, 0);
#else
  __builtin_amdgcn_tensor_load_to_lds(g0, g1, z4, z4, 0);
#endif
  __builtin_amdgcn_s_wait_tensorcnt(0);
}
#endif

// ---------------------------------------------------------------------------
// Kernel 0: fold u into edge/node biases; swizzle We[0:384] / Wv[0:256] into
// bf16 B-fragment layout:  idx = ((chunk*8 + ntile)*32 + lane)*8 + v
//   packed pair K = chunk*32 + (lane>>4)*16 + 2v (,+1), col N = ntile*16 + (lane&15)
// ---------------------------------------------------------------------------
__global__ __launch_bounds__(256) void gn_prep_kernel(
    const float* __restrict__ We, const float* __restrict__ be,
    const float* __restrict__ Wv, const float* __restrict__ bv,
    const float* __restrict__ u,
    unsigned* __restrict__ We_swz, unsigned* __restrict__ Wv_swz,
    float* __restrict__ c_e, float* __restrict__ c_v)
{
  const int total = 24576 + 16384 + 256;
  for (int i = blockIdx.x * blockDim.x + threadIdx.x; i < total;
       i += gridDim.x * blockDim.x) {
    if (i < 24576) {
      int v = i & 7, lane = (i >> 3) & 31, t = (i >> 8) & 7, chunk = i >> 11;
      int k = chunk * 32 + (lane >> 4) * 16 + v * 2;
      int n = t * 16 + (lane & 15);
      We_swz[i] = pack_bf16x2(We[k * DD + n], We[(k + 1) * DD + n]);
    } else if (i < 24576 + 16384) {
      int j = i - 24576;
      int v = j & 7, lane = (j >> 3) & 31, t = (j >> 8) & 7, chunk = j >> 11;
      int k = chunk * 32 + (lane >> 4) * 16 + v * 2;
      int n = t * 16 + (lane & 15);
      Wv_swz[j] = pack_bf16x2(Wv[k * DD + n], Wv[(k + 1) * DD + n]);
    } else {
      int j = i - 40960;
      if (j < 128) {
        float s = be[j];
        for (int k = 0; k < 128; ++k) s += u[k] * We[(384 + k) * DD + j];
        c_e[j] = s;
      } else {
        j -= 128;
        float s = bv[j];
        for (int k = 0; k < 128; ++k) s += u[k] * Wv[(256 + k) * DD + j];
        c_v[j] = s;
      }
    }
  }
}

// ---------------------------------------------------------------------------
// Kernel 1: edge block. 8 waves/block, 16 edges/wave, full 128 outputs/wave.
// K = 384 (edge_attr | node[recv] | node[send]); 12 chunks x 8 ntiles WMMAs.
// Weights staged L2 -> LDS by the Tensor Data Mover (one D# per block).
// ---------------------------------------------------------------------------
__global__ __launch_bounds__(256) void gn_edge_kernel(
    const float* __restrict__ edge_attr,
    const float* __restrict__ node_attr,
    const int*   __restrict__ senders,
    const int*   __restrict__ receivers,
    const unsigned* __restrict__ We_swz,
    const float* __restrict__ c_e,
    float* __restrict__ e_new,
    float* __restrict__ aggr_e,
    float* __restrict__ esum)
{
  extern __shared__ unsigned char smem[];
  unsigned* lw   = (unsigned*)smem;               // 24576 u32 = 96 KB weights
  float*    lcol = (float*)(smem + 24576u * 4u);  // 128 f32 column sums

  const int tid = threadIdx.x;
#if USE_TDM
  if (tid < 32) {  // one wave issues the TDM transfer and waits on TENSORcnt
    tdm_stage_1d(We_swz, (unsigned)__builtin_amdgcn_groupstaticsize(), 12288u);
  }
#else
  {
    const uint4* g = (const uint4*)We_swz;
    uint4*       l = (uint4*)lw;
    #pragma unroll 4
    for (int i = tid; i < 6144; i += 256) l[i] = g[i];
  }
#endif
  if (tid < 128) lcol[tid] = 0.0f;
  __syncthreads();

  const int lane = tid & 31;
  const int wave = tid >> 5;
  const int m    = lane & 15;
  const int half = lane >> 4;

  const long long tile_row0 = (long long)blockIdx.x * 128 + wave * 16;
  const long long arow      = tile_row0 + m;

  const float* s0 = edge_attr + arow * DD;
  const float* s1 = node_attr + (long long)receivers[arow] * DD;
  const float* s2 = node_attr + (long long)senders[arow] * DD;

  v8f acc[8];
  #pragma unroll
  for (int t = 0; t < 8; ++t) {
    const float bias = c_e[t * 16 + m];
    #pragma unroll
    for (int r = 0; r < 8; ++r) acc[t][r] = bias;
  }

  #pragma unroll
  for (int chunk = 0; chunk < 12; ++chunk) {
    const float* src = (chunk < 4) ? (s0 + (chunk << 5))
                     : (chunk < 8) ? (s1 + ((chunk - 4) << 5))
                                   : (s2 + ((chunk - 8) << 5));
    v16bf afrag;
    #pragma unroll
    for (int v = 0; v < 8; ++v) {   // A frag: K = half*16 + v*2
      const int kl = (half << 4) + (v << 1);
      afrag[2 * v]     = (__bf16)src[kl];
      afrag[2 * v + 1] = (__bf16)src[kl + 1];
    }

    #pragma unroll
    for (int t = 0; t < 8; ++t) {
      const uint4* p = (const uint4*)(lw + ((((chunk << 3) + t) << 5) + lane) * 8);
      const uint4 q0 = p[0];
      const uint4 q1 = p[1];
      v8u bfu;
      bfu[0] = q0.x; bfu[1] = q0.y; bfu[2] = q0.z; bfu[3] = q0.w;
      bfu[4] = q1.x; bfu[5] = q1.y; bfu[6] = q1.z; bfu[7] = q1.w;
      const v16bf bfrag = __builtin_bit_cast(v16bf, bfu);
      acc[t] = wmma_bf16(afrag, bfrag, acc[t]);
    }
  }

  // Epilogue: ReLU, store e_new, scatter-add into aggr_e, column sums.
  const long long rowbase = tile_row0 + half * 8;   // C: row = r + 8*half
  int rr[8];
  #pragma unroll
  for (int r = 0; r < 8; ++r) rr[r] = receivers[rowbase + r];

  #pragma unroll
  for (int t = 0; t < 8; ++t) {
    const int col = t * 16 + m;
    float psum = 0.0f;
    #pragma unroll
    for (int r = 0; r < 8; ++r) {
      float v = acc[t][r];
      v = v > 0.0f ? v : 0.0f;
      e_new[(rowbase + r) * DD + col] = v;
      atomicAdd(&aggr_e[(long long)rr[r] * DD + col], v);
      psum += v;
    }
    atomicAdd(&lcol[col], psum);
  }
  __syncthreads();
  if (tid < 128) atomicAdd(&esum[tid], lcol[tid]);
}

// ---------------------------------------------------------------------------
// Kernel 2: node block. K = 256 (aggr_e | node_attr); 8 chunks x 8 ntiles.
// ---------------------------------------------------------------------------
__global__ __launch_bounds__(256) void gn_node_kernel(
    const float* __restrict__ aggr_e,
    const float* __restrict__ node_attr,
    const unsigned* __restrict__ Wv_swz,
    const float* __restrict__ c_v,
    float* __restrict__ v_new,
    float* __restrict__ vsum,
    int N)
{
  extern __shared__ unsigned char smem[];
  unsigned* lw   = (unsigned*)smem;               // 16384 u32 = 64 KB
  float*    lcol = (float*)(smem + 16384u * 4u);

  const int tid = threadIdx.x;
#if USE_TDM
  if (tid < 32) {
    tdm_stage_1d(Wv_swz, (unsigned)__builtin_amdgcn_groupstaticsize(), 8192u);
  }
#else
  {
    const uint4* g = (const uint4*)Wv_swz;
    uint4*       l = (uint4*)lw;
    #pragma unroll 4
    for (int i = tid; i < 4096; i += 256) l[i] = g[i];
  }
#endif
  if (tid < 128) lcol[tid] = 0.0f;
  __syncthreads();

  const int lane = tid & 31;
  const int wave = tid >> 5;
  const int m    = lane & 15;
  const int half = lane >> 4;

  const long long tile_row0 = (long long)blockIdx.x * 128 + wave * 16;
  const long long arow      = tile_row0 + m;
  const long long arowc     = (arow < N) ? arow : 0;

  const float* s0 = aggr_e    + arowc * DD;
  const float* s1 = node_attr + arowc * DD;

  v8f acc[8];
  #pragma unroll
  for (int t = 0; t < 8; ++t) {
    const float bias = c_v[t * 16 + m];
    #pragma unroll
    for (int r = 0; r < 8; ++r) acc[t][r] = bias;
  }

  #pragma unroll
  for (int chunk = 0; chunk < 8; ++chunk) {
    const float* src = (chunk < 4) ? (s0 + (chunk << 5))
                                   : (s1 + ((chunk - 4) << 5));
    v16bf afrag;
    #pragma unroll
    for (int v = 0; v < 8; ++v) {
      const int kl = (half << 4) + (v << 1);
      afrag[2 * v]     = (__bf16)src[kl];
      afrag[2 * v + 1] = (__bf16)src[kl + 1];
    }

    #pragma unroll
    for (int t = 0; t < 8; ++t) {
      const uint4* p = (const uint4*)(lw + ((((chunk << 3) + t) << 5) + lane) * 8);
      const uint4 q0 = p[0];
      const uint4 q1 = p[1];
      v8u bfu;
      bfu[0] = q0.x; bfu[1] = q0.y; bfu[2] = q0.z; bfu[3] = q0.w;
      bfu[4] = q1.x; bfu[5] = q1.y; bfu[6] = q1.z; bfu[7] = q1.w;
      const v16bf bfrag = __builtin_bit_cast(v16bf, bfu);
      acc[t] = wmma_bf16(afrag, bfrag, acc[t]);
    }
  }

  const long long rowbase = tile_row0 + half * 8;
  #pragma unroll
  for (int t = 0; t < 8; ++t) {
    const int col = t * 16 + m;
    float psum = 0.0f;
    #pragma unroll
    for (int r = 0; r < 8; ++r) {
      const long long grow = rowbase + r;
      float v = acc[t][r];
      v = v > 0.0f ? v : 0.0f;
      if (grow < N) {
        v_new[grow * DD + col] = v;
        psum += v;
      }
    }
    atomicAdd(&lcol[col], psum);
  }
  __syncthreads();
  if (tid < 128) atomicAdd(&vsum[tid], lcol[tid]);
}

// ---------------------------------------------------------------------------
// Kernel 3: global block. [3D] @ [3D, D] -- trivial, scalar f32.
// ---------------------------------------------------------------------------
__global__ void gn_global_kernel(
    const float* __restrict__ esum, const float* __restrict__ vsum,
    const float* __restrict__ u,    const float* __restrict__ Wu,
    const float* __restrict__ bu,   float* __restrict__ u_new)
{
  const int j = threadIdx.x;  // 128 threads
  const float invE = 1.0f / (float)NEDGES;
  const float invN = 1.0f / (float)NNODES;
  float s = bu[j];
  for (int k = 0; k < 128; ++k) s += (esum[k] * invE) * Wu[k * DD + j];
  for (int k = 0; k < 128; ++k) s += (vsum[k] * invN) * Wu[(128 + k) * DD + j];
  for (int k = 0; k < 128; ++k) s += u[k] * Wu[(256 + k) * DD + j];
  u_new[j] = s > 0.0f ? s : 0.0f;
}

// ---------------------------------------------------------------------------
// Workspace layout (bytes):
//   [0,       98304)   We_swz   (bf16 pairs, 384x128)
//   [98304,  163840)   Wv_swz   (bf16 pairs, 256x128)
//   [163840, 164352)   c_e      (128 f32)
//   [164352, 164864)   c_v      (128 f32)
//   [164864, 165376)   esum     (128 f32, zeroed each call)
//   [165376, 165888)   vsum     (128 f32, zeroed each call)
//   [165888, 10405888) aggr_e   (20000x128 f32, zeroed each call)
// ---------------------------------------------------------------------------
extern "C" void kernel_launch(void* const* d_in, const int* in_sizes, int n_in,
                              void* d_out, int out_size, void* d_ws, size_t ws_size,
                              hipStream_t stream) {
  const float* edge_attr = (const float*)d_in[0];
  const float* node_attr = (const float*)d_in[1];
  const float* u         = (const float*)d_in[2];
  const int*   senders   = (const int*)d_in[3];
  const int*   receivers = (const int*)d_in[4];
  const float* We        = (const float*)d_in[5];
  const float* be        = (const float*)d_in[6];
  const float* Wv        = (const float*)d_in[7];
  const float* bv        = (const float*)d_in[8];
  const float* Wu        = (const float*)d_in[9];
  const float* bu        = (const float*)d_in[10];

  float* out   = (float*)d_out;
  float* e_new = out;
  float* v_new = out + (size_t)NEDGES * DD;
  float* u_new = v_new + (size_t)NNODES * DD;

  char* ws = (char*)d_ws;
  unsigned* We_swz = (unsigned*)(ws);
  unsigned* Wv_swz = (unsigned*)(ws + 98304);
  float* c_e  = (float*)(ws + 163840);
  float* c_v  = (float*)(ws + 164352);
  float* esum = (float*)(ws + 164864);
  float* vsum = (float*)(ws + 165376);
  float* aggr = (float*)(ws + 165888);

  // Zero the accumulators (esum | vsum | aggr_e are contiguous).
  hipMemsetAsync(ws + 164864, 0, 1024 + (size_t)NNODES * DD * sizeof(float), stream);

  // Allow >64KB dynamic LDS (WGP has 320KB).
  (void)hipFuncSetAttribute((const void*)gn_edge_kernel,
                            hipFuncAttributeMaxDynamicSharedMemorySize, 24576 * 4 + 512);
  (void)hipFuncSetAttribute((const void*)gn_node_kernel,
                            hipFuncAttributeMaxDynamicSharedMemorySize, 16384 * 4 + 512);

  gn_prep_kernel<<<64, 256, 0, stream>>>(We, be, Wv, bv, u, We_swz, Wv_swz, c_e, c_v);

  // 320000 edges / (8 waves * 16 edges) = 2500 blocks, exact.
  gn_edge_kernel<<<2500, 256, 24576 * 4 + 512, stream>>>(
      edge_attr, node_attr, senders, receivers, We_swz, c_e, e_new, aggr, esum);

  // ceil(20000 / 128) = 157 blocks.
  gn_node_kernel<<<157, 256, 16384 * 4 + 512, stream>>>(
      aggr, node_attr, Wv_swz, c_v, v_new, vsum, NNODES);

  gn_global_kernel<<<1, 128, 0, stream>>>(esum, vsum, u, Wu, bu, u_new);
}
